// CausalTradeGNN_3015067042046
// MI455X (gfx1250) — matrix operand order repaired and checked by reading, not compiled, not run on hardware
//
#include <hip/hip_runtime.h>
#include <hip/hip_bf16.h>
#include <math.h>

typedef __attribute__((ext_vector_type(16))) _Float16 v16h;
typedef __attribute__((ext_vector_type(8)))  _Float16 v8h;
typedef __attribute__((ext_vector_type(8)))  float    v8f;

#define BN_EPS 1e-5f
#define MLP_KP 160   // 139 padded to 5 K-steps of 32
#define MLP_H1 128
#define MLP_H2 64

// ---------------------------------------------------------------- helpers
__device__ __forceinline__ float atomicMaxFloat(float* addr, float value) {
  if (value >= 0.f)
    return __int_as_float(atomicMax((int*)addr, __float_as_int(value)));
  else
    return __uint_as_float(atomicMin((unsigned int*)addr, __float_as_uint(value)));
}

__device__ __forceinline__ v16h load_a_frag(const _Float16* p) {
  // lane's A elements: halves [0..7] and [16..23] relative to p (two 16B groups)
  v8h lo = *(const v8h*)p;
  v8h hi = *(const v8h*)(p + 16);
  return __builtin_shufflevector(lo, hi, 0, 1, 2, 3, 4, 5, 6, 7,
                                 8, 9, 10, 11, 12, 13, 14, 15);
}

// f32 [R,C] -> f16 [Rp,Cp] row-major, zero padded (activations)
__global__ void cvt_pad_f16(const float* __restrict__ in, _Float16* __restrict__ out,
                            int R, int C, int Rp, int Cp) {
  int t = blockIdx.x * blockDim.x + threadIdx.x;
  if (t >= Rp * Cp) return;
  int r = t / Cp, c = t % Cp;
  float v = (r < R && c < C) ? in[(size_t)r * C + c] : 0.f;
  out[t] = (_Float16)v;
}

// f32 W[R,Nout] -> f16 packed in WMMA B-fragment order, K zero-padded to Kp.
// layout: [kt][nt][lane][16] halves, kt = K/32 tiles, nt = Nout/16 tiles.
// lane: n = nt*16 + (lane&15); k = kt*32 + (j>>3)*16 + (lane>>4)*8 + (j&7).
__global__ void cvt_pack_b_f16(const float* __restrict__ W, _Float16* __restrict__ out,
                               int R, int Nout, int Kp) {
  int t = blockIdx.x * blockDim.x + threadIdx.x;
  if (t >= Kp * Nout) return;
  int j      = t & 15;
  int lane   = (t >> 4) & 31;
  int tileid = t >> 9;
  int NT     = Nout >> 4;
  int nt     = tileid % NT;
  int kt     = tileid / NT;
  int n = nt * 16 + (lane & 15);
  int k = kt * 32 + (j >> 3) * 16 + (lane >> 4) * 8 + (j & 7);
  out[t] = (_Float16)((k < R) ? W[(size_t)k * Nout + n] : 0.f);
}

__global__ void init_softmax(float* __restrict__ amax, float* __restrict__ denom, int n) {
  int t = blockIdx.x * blockDim.x + threadIdx.x;
  if (t >= n) return;
  amax[t] = -INFINITY;
  denom[t] = 0.f;
}

// ------------------------------------------------------- generic WMMA GEMM
// C[M,N] = A_f16[M,K] @ Bpacked_f16 + bias ; one 16x16 tile per wave32
__global__ __launch_bounds__(128)
void wmma_gemm_f16(const _Float16* __restrict__ A, const _Float16* __restrict__ Bp,
                   const float* __restrict__ bias, float* __restrict__ C,
                   int M, int K, int N) {
  int wave = blockIdx.x * (blockDim.x >> 5) + (threadIdx.x >> 5);
  int ntiles = N >> 4;
  int mt = wave / ntiles;
  int nt = wave - mt * ntiles;
  if (mt * 16 >= M) return;                    // wave-uniform: EXEC stays all-ones
  int lane = threadIdx.x & 31;
  int m    = lane & 15;                        // A row within tile == C column
  int ng   = lane >> 4;
  int row  = mt * 16 + m;
  int n0   = nt * 16;

  v8f acc = {};
  for (int k0 = 0; k0 < K; k0 += 32) {
    v16h a = load_a_frag(A + (size_t)row * K + k0 + ng * 8);
    const _Float16* Br = Bp + (((size_t)(k0 >> 5) * ntiles + nt) * 32 + lane) * 16;
    v16h b = *(const v16h*)Br;                 // 32B contiguous -> 2x b128
    acc = __builtin_amdgcn_wmma_f32_16x16x32_f16(false, a, false, b, (short)0, acc, false, false);
  }
  float bv = bias ? bias[n0 + m] : 0.f;
#pragma unroll
  for (int r = 0; r < 8; ++r) {
    int rr = mt * 16 + r + 8 * ng;
    C[(size_t)rr * N + n0 + m] = acc[r] + bv;
  }
}

// ------------------------------------------------------------ edge kernels
__global__ void gravity_kernel(const float* __restrict__ x, const int* __restrict__ ei,
                               const float* __restrict__ ea, const float* __restrict__ W,
                               const float* __restrict__ b, float* __restrict__ grav, int E) {
  int e = blockIdx.x * blockDim.x + threadIdx.x;
  if (e >= E) return;
  int r = ei[e], c = ei[E + e];
  float acc = b[0];
#pragma unroll
  for (int j = 0; j < 4; ++j) acc += x[r * 4 + j] * W[j];
#pragma unroll
  for (int j = 0; j < 4; ++j) acc += x[c * 4 + j] * W[4 + j];
#pragma unroll
  for (int j = 0; j < 10; ++j) acc += ea[(size_t)e * 10 + j] * W[8 + j];
  grav[e] = acc;
}

// pass A: alpha = q[col] . (K[row] + ea@We) / sqrt(C) ; atomic segment max over col
__global__ void edge_alpha(const int* __restrict__ ei, const float* __restrict__ ea,
                           const float* __restrict__ We, const float* __restrict__ Q,
                           const float* __restrict__ Kb, float* __restrict__ alpha,
                           float* __restrict__ amax, int E, int H, int C, int HC,
                           float inv_sqrt_c) {
  int t = blockIdx.x * blockDim.x + threadIdx.x;
  if (t >= E * H) return;
  int e = t / H, h = t - e * H;
  int r = ei[e], c = ei[E + e];
  const float* q = Q  + (size_t)c * HC + h * C;
  const float* k = Kb + (size_t)r * HC + h * C;
  float ae[10];
#pragma unroll
  for (int j = 0; j < 10; ++j) ae[j] = ea[(size_t)e * 10 + j];
  float acc = 0.f;
  for (int cc = 0; cc < C; ++cc) {
    float ev = 0.f;
#pragma unroll
    for (int j = 0; j < 10; ++j) ev += ae[j] * We[j * HC + h * C + cc];
    acc += q[cc] * (k[cc] + ev);
  }
  acc *= inv_sqrt_c;
  alpha[t] = acc;
  atomicMaxFloat(&amax[c * H + h], acc);
}

// pass B: ex = exp(alpha - amax[col]); denom[col] += ex
__global__ void edge_expsum(const int* __restrict__ ei, float* __restrict__ alpha,
                            const float* __restrict__ amax, float* __restrict__ denom,
                            int E, int H) {
  int t = blockIdx.x * blockDim.x + threadIdx.x;
  if (t >= E * H) return;
  int e = t / H, h = t - e * H;
  int c = ei[E + e];
  float ex = expf(alpha[t] - amax[c * H + h]);
  alpha[t] = ex;
  atomicAdd(&denom[c * H + h], ex);
}

// pass C: OUT[col] += (ex/denom) * (V[row] + ea@We)
__global__ void edge_scatter(const int* __restrict__ ei, const float* __restrict__ ea,
                             const float* __restrict__ We, const float* __restrict__ V,
                             const float* __restrict__ alpha, const float* __restrict__ denom,
                             float* __restrict__ OUT, int E, int H, int C, int HC) {
  long t = (long)blockIdx.x * blockDim.x + threadIdx.x;
  if (t >= (long)E * HC) return;
  int e  = (int)(t / HC);
  int hc = (int)(t - (long)e * HC);
  int h  = hc / C;
  int r = ei[e], c = ei[E + e];
  float w = alpha[(size_t)e * H + h] / (denom[c * H + h] + 1e-16f);
  float ev = 0.f;
#pragma unroll
  for (int j = 0; j < 10; ++j) ev += ea[(size_t)e * 10 + j] * We[j * HC + hc];
  float v = V[(size_t)r * HC + hc] + ev;
  atomicAdd(&OUT[(size_t)c * HC + hc], w * v);
}

// ---------------------------------------------------------------- batchnorm
__global__ __launch_bounds__(256)
void bn_stats(const float* __restrict__ X, float* __restrict__ mean, float* __restrict__ var,
              int Nn, int D) {
  int d = blockIdx.x;
  float s = 0.f, s2 = 0.f;
  for (int n = threadIdx.x; n < Nn; n += 256) {
    float v = X[(size_t)n * D + d];
    s += v; s2 += v * v;
  }
#pragma unroll
  for (int off = 16; off > 0; off >>= 1) {
    s  += __shfl_down(s, off);
    s2 += __shfl_down(s2, off);
  }
  __shared__ float rs[8], rs2[8];
  int wid = threadIdx.x >> 5;
  if ((threadIdx.x & 31) == 0) { rs[wid] = s; rs2[wid] = s2; }
  __syncthreads();
  if (threadIdx.x == 0) {
    float ts = 0.f, ts2 = 0.f;
    for (int i = 0; i < 8; ++i) { ts += rs[i]; ts2 += rs2[i]; }
    float m = ts / Nn;
    mean[d] = m;
    var[d]  = ts2 / Nn - m * m;
  }
}

__global__ void bn_apply_elu(float* __restrict__ X, const float* __restrict__ mean,
                             const float* __restrict__ var, const float* __restrict__ g,
                             const float* __restrict__ b, int Nn, int D) {
  long t = (long)blockIdx.x * blockDim.x + threadIdx.x;
  if (t >= (long)Nn * D) return;
  int d = (int)(t % D);
  float v = (X[t] - mean[d]) * rsqrtf(var[d] + BN_EPS) * g[d] + b[d];
  X[t] = v > 0.f ? v : expf(v) - 1.f;
}

// ------------------------------------------------- fused per-edge MLP (WMMA)
// emb[E,139] -> relu(@W1+b1)[128] -> relu(@W2+b2)[64] -> @W3+b3 -> pred[E]
__global__ __launch_bounds__(256)
void mlp_fused(const float* __restrict__ h, const int* __restrict__ ei,
               const float* __restrict__ ea, const float* __restrict__ grav,
               const float* __restrict__ W1, const float* __restrict__ b1,
               const float* __restrict__ W2, const float* __restrict__ b2,
               const float* __restrict__ W3, const float* __restrict__ b3,
               float* __restrict__ pred, int E) {
  // weights staged in WMMA B-fragment order: [(kt*NT+nt)*32 + lane][16]
  __shared__ _Float16 sW1p[MLP_KP * MLP_H1];
  __shared__ _Float16 sW2p[MLP_H1 * MLP_H2];
  __shared__ float    sW3[MLP_H2], sB1[MLP_H1], sB2[MLP_H2];
  __shared__ _Float16 sEmb[8][16 * MLP_KP];
  __shared__ _Float16 sZ1[8][16 * MLP_H1];
  __shared__ _Float16 sZ2[8][16 * MLP_H2];

  int tid = threadIdx.x;
  for (int i = tid; i < MLP_KP * MLP_H1; i += 256) {      // NT=8
    int j = i & 15, ln = (i >> 4) & 31, tileid = i >> 9;
    int n = (tileid % 8) * 16 + (ln & 15);
    int k = (tileid / 8) * 32 + (j >> 3) * 16 + (ln >> 4) * 8 + (j & 7);
    sW1p[i] = (_Float16)((k < 139) ? W1[k * MLP_H1 + n] : 0.f);
  }
  for (int i = tid; i < MLP_H1 * MLP_H2; i += 256) {      // NT=4
    int j = i & 15, ln = (i >> 4) & 31, tileid = i >> 9;
    int n = (tileid % 4) * 16 + (ln & 15);
    int k = (tileid / 4) * 32 + (j >> 3) * 16 + (ln >> 4) * 8 + (j & 7);
    sW2p[i] = (_Float16)W2[k * MLP_H2 + n];
  }
  if (tid < MLP_H2) { sW3[tid] = W3[tid]; sB2[tid] = b2[tid]; }
  if (tid < MLP_H1) sB1[tid] = b1[tid];

  int wid  = tid >> 5, lane = tid & 31;
  int e0   = (blockIdx.x * 8 + wid) * 16;
  _Float16* emb = sEmb[wid];
  _Float16* z1  = sZ1[wid];
  _Float16* z2  = sZ2[wid];

  // stage 16x160 emb tile (zero padded cols 139..159 and edges >= E)
  for (int i = lane; i < 16 * MLP_KP; i += 32) {
    int rrow = i / MLP_KP, c = i - rrow * MLP_KP;
    int e = e0 + rrow;
    float v = 0.f;
    if (e < E) {
      if (c < 64)        { int rn = ei[e];     v = h[(size_t)rn * 64 + c]; }
      else if (c < 128)  { int cn = ei[E + e]; v = h[(size_t)cn * 64 + (c - 64)]; }
      else if (c < 138)  v = ea[(size_t)e * 10 + (c - 128)];
      else if (c == 138) v = grav[e];
    }
    emb[i] = (_Float16)v;
  }
  __syncthreads();

  int m = lane & 15, ng = lane >> 4;

  // GEMM1: [16,160] x [160,128]
  for (int nt = 0; nt < MLP_H1 / 16; ++nt) {
    int n0 = nt * 16;
    v8f acc = {};
    for (int k0 = 0; k0 < MLP_KP; k0 += 32) {
      v16h a = load_a_frag(emb + m * MLP_KP + k0 + ng * 8);
      v16h b = *(const v16h*)(sW1p + (((k0 >> 5) * 8 + nt) * 32 + lane) * 16);
      acc = __builtin_amdgcn_wmma_f32_16x16x32_f16(false, a, false, b, (short)0, acc, false, false);
    }
    float bv = sB1[n0 + m];
#pragma unroll
    for (int r = 0; r < 8; ++r) {
      float v = acc[r] + bv;
      z1[(r + 8 * ng) * MLP_H1 + n0 + m] = (_Float16)(v > 0.f ? v : 0.f);
    }
  }
  __syncthreads();

  // GEMM2: [16,128] x [128,64]
  for (int nt = 0; nt < MLP_H2 / 16; ++nt) {
    int n0 = nt * 16;
    v8f acc = {};
    for (int k0 = 0; k0 < MLP_H1; k0 += 32) {
      v16h a = load_a_frag(z1 + m * MLP_H1 + k0 + ng * 8);
      v16h b = *(const v16h*)(sW2p + (((k0 >> 5) * 4 + nt) * 32 + lane) * 16);
      acc = __builtin_amdgcn_wmma_f32_16x16x32_f16(false, a, false, b, (short)0, acc, false, false);
    }
    float bv = sB2[n0 + m];
#pragma unroll
    for (int r = 0; r < 8; ++r) {
      float v = acc[r] + bv;
      z2[(r + 8 * ng) * MLP_H2 + n0 + m] = (_Float16)(v > 0.f ? v : 0.f);
    }
  }
  __syncthreads();

  // final dot: each lane handles half a row, combine across lane^16
  float p = 0.f;
  for (int c = ng * 32; c < ng * 32 + 32; ++c) p += (float)z2[m * MLP_H2 + c] * sW3[c];
  p += __shfl_xor(p, 16);
  if (ng == 0) {
    int e = e0 + m;
    if (e < E) pred[e] = p + b3[0];
  }
}

// =========================================================== orchestration
extern "C" void kernel_launch(void* const* d_in, const int* in_sizes, int n_in,
                              void* d_out, int out_size, void* d_ws, size_t ws_size,
                              hipStream_t stream) {
  const int N = in_sizes[0] / 4;   // 20000
  const int E = in_sizes[1] / 2;   // 200000

  const float* x  = (const float*)d_in[0];
  const int*   ei = (const int*)d_in[1];
  const float* ea = (const float*)d_in[2];
  const float* gW = (const float*)d_in[3];
  const float* gB = (const float*)d_in[4];
  const float* mlpW1 = (const float*)d_in[38];
  const float* mlpB1 = (const float*)d_in[39];
  const float* mlpW2 = (const float*)d_in[40];
  const float* mlpB2 = (const float*)d_in[41];
  const float* mlpW3 = (const float*)d_in[42];
  const float* mlpB3 = (const float*)d_in[43];

  // workspace layout
  size_t off = 0;
  auto alloc = [&](size_t bytes) {
    void* p = (char*)d_ws + off;
    off += (bytes + 255) & ~(size_t)255;
    return p;
  };
  float*    grav  = (float*)alloc((size_t)E * 4);
  float*    bufA  = (float*)alloc((size_t)N * 256 * 4);
  float*    bufB  = (float*)alloc((size_t)N * 256 * 4);
  float*    Qb    = (float*)alloc((size_t)N * 256 * 4);
  float*    Kb    = (float*)alloc((size_t)N * 256 * 4);
  float*    Vb    = (float*)alloc((size_t)N * 256 * 4);
  _Float16* hf16  = (_Float16*)alloc((size_t)N * 256 * 2);
  _Float16* wq16  = (_Float16*)alloc(256 * 256 * 2);
  _Float16* wk16  = (_Float16*)alloc(256 * 256 * 2);
  _Float16* wv16  = (_Float16*)alloc(256 * 256 * 2);
  _Float16* ws16  = (_Float16*)alloc(256 * 256 * 2);
  float*    amax  = (float*)alloc((size_t)N * 4 * 4);
  float*    denom = (float*)alloc((size_t)N * 4 * 4);
  float*    alpha = (float*)alloc((size_t)E * 4 * 4);
  float*    mean  = (float*)alloc(256 * 4);
  float*    var   = (float*)alloc(256 * 4);

  auto cdiv = [](long a, long b) { return (int)((a + b - 1) / b); };

  // gravity scalar per edge
  gravity_kernel<<<cdiv(E, 256), 256, 0, stream>>>(x, ei, ea, gW, gB, grav, E);

  const int base_[3] = {5, 16, 27};
  const int cin_[3]  = {4, 256, 256};
  const int kpad_[3] = {32, 256, 256};
  const int H_[3]    = {4, 4, 1};
  const int C_ = 64;

  const float* h_in = x;
  float* h_out_ptrs[3] = {bufA, bufB, bufA};

  for (int l = 0; l < 3; ++l) {
    const int cin = cin_[l], Kp = kpad_[l], H = H_[l], HC = H * C_;
    const int b0 = base_[l];
    const float* Wq = (const float*)d_in[b0 + 0];
    const float* bq = (const float*)d_in[b0 + 1];
    const float* Wk = (const float*)d_in[b0 + 2];
    const float* bk = (const float*)d_in[b0 + 3];
    const float* Wv = (const float*)d_in[b0 + 4];
    const float* bv = (const float*)d_in[b0 + 5];
    const float* We = (const float*)d_in[b0 + 6];
    const float* Ws = (const float*)d_in[b0 + 7];
    const float* bs = (const float*)d_in[b0 + 8];
    const float* bg = (const float*)d_in[b0 + 9];
    const float* bb = (const float*)d_in[b0 + 10];
    float* OUT = h_out_ptrs[l];

    // f16 conversions: activations row-major padded; weights packed to fragment order
    cvt_pad_f16<<<cdiv((long)N * Kp, 256), 256, 0, stream>>>(h_in, hf16, N, cin, N, Kp);
    cvt_pack_b_f16<<<cdiv((long)Kp * HC, 256), 256, 0, stream>>>(Wq, wq16, cin, HC, Kp);
    cvt_pack_b_f16<<<cdiv((long)Kp * HC, 256), 256, 0, stream>>>(Wk, wk16, cin, HC, Kp);
    cvt_pack_b_f16<<<cdiv((long)Kp * HC, 256), 256, 0, stream>>>(Wv, wv16, cin, HC, Kp);
    cvt_pack_b_f16<<<cdiv((long)Kp * HC, 256), 256, 0, stream>>>(Ws, ws16, cin, HC, Kp);

    // node projections via WMMA (one 16x16 tile per wave, 4 waves/block)
    int tiles = (N / 16) * (HC / 16);
    int gblk  = cdiv(tiles, 4);
    wmma_gemm_f16<<<gblk, 128, 0, stream>>>(hf16, wq16, bq, Qb, N, Kp, HC);
    wmma_gemm_f16<<<gblk, 128, 0, stream>>>(hf16, wk16, bk, Kb, N, Kp, HC);
    wmma_gemm_f16<<<gblk, 128, 0, stream>>>(hf16, wv16, bv, Vb, N, Kp, HC);
    wmma_gemm_f16<<<gblk, 128, 0, stream>>>(hf16, ws16, bs, OUT, N, Kp, HC);  // skip init

    // segment softmax attention (3 edge passes)
    init_softmax<<<cdiv((long)N * H, 256), 256, 0, stream>>>(amax, denom, N * H);
    edge_alpha<<<cdiv((long)E * H, 256), 256, 0, stream>>>(ei, ea, We, Qb, Kb, alpha, amax,
                                                           E, H, C_, HC, 0.125f);
    edge_expsum<<<cdiv((long)E * H, 256), 256, 0, stream>>>(ei, alpha, amax, denom, E, H);
    edge_scatter<<<cdiv((long)E * HC, 256), 256, 0, stream>>>(ei, ea, We, Vb, alpha, denom,
                                                              OUT, E, H, C_, HC);

    // batchnorm + elu (in place)
    bn_stats<<<HC, 256, 0, stream>>>(OUT, mean, var, N, HC);
    bn_apply_elu<<<cdiv((long)N * HC, 256), 256, 0, stream>>>(OUT, mean, var, bg, bb, N, HC);

    h_in = OUT;
  }

  // fused per-edge MLP: 8 waves/block, 16 edges/wave
  int mlp_tiles = cdiv(E, 16);
  mlp_fused<<<cdiv(mlp_tiles, 8), 256, 0, stream>>>(h_out_ptrs[2], ei, ea, grav,
                                                    mlpW1, mlpB1, mlpW2, mlpB2, mlpW3, mlpB3,
                                                    (float*)d_out, E);
}
